// MyModel_60138132079051
// MI455X (gfx1250) — matrix operand order repaired and checked by reading, not compiled, not run on hardware
//
#include <hip/hip_runtime.h>
#include <hip/hip_bf16.h>

typedef __attribute__((ext_vector_type(16))) _Float16 v16h;
typedef __attribute__((ext_vector_type(8)))  float    v8f;

#define DIMC 128
#define PW   8
#define HH   256
#define WW   256
#define NB   2
#define S1   32
#define S2   32
#define NWIN 2048              // NB*S1*S2
#define SCALE_F 0.08838834764831845f   // 128^-0.5

// ---------------------------------------------------------------------------
// weights fp32 (co,ci,3,3) -> f16 [tap][co][ci]
__global__ __launch_bounds__(256) void pack_w_kernel(const float* __restrict__ w,
                                                     _Float16* __restrict__ wh) {
    int idx = blockIdx.x * 256 + threadIdx.x;            // < 9*128*128 = 147456
    if (idx >= 9 * DIMC * DIMC) return;
    int ci  = idx & 127;
    int co  = (idx >> 7) & 127;
    int tap = idx >> 14;
    wh[idx] = (_Float16)w[(co * DIMC + ci) * 9 + tap];
}

// ---------------------------------------------------------------------------
// 3x3 conv, implicit GEMM via WMMA.  in: fp32 NCHW (2,128,256,256)
// out: f16 NHWC (2,256,256,128).  WG: 256 thr = 8 waves, tile 128cout x 16pix.
__global__ __launch_bounds__(256) void conv3x3_kernel(const float* __restrict__ in,
                                                      const _Float16* __restrict__ wh,
                                                      const float* __restrict__ bias,
                                                      _Float16* __restrict__ out) {
    __shared__ __align__(16) _Float16 s_in[3 * 18 * DIMC];   // [(r*18+c)*128+ci]

    const int t  = threadIdx.x;
    const int w0 = blockIdx.x * 16;
    const int h  = blockIdx.y;
    const int b  = blockIdx.z;

    // stage 3x18 x 128ch patch (zero-padded borders), fp32 -> f16
    #pragma unroll
    for (int k = 0; k < 27; ++k) {
        int idx = k * 256 + t;                 // 0..6911
        int ci  = idx / 54;
        int pos = idx % 54;
        int r = pos / 18, c = pos % 18;
        int hh = h + r - 1, ww = w0 + c - 1;
        float v = 0.f;
        if (hh >= 0 && hh < HH && ww >= 0 && ww < WW)
            v = in[(((size_t)b * DIMC + ci) * HH + hh) * WW + ww];
        s_in[(r * 18 + c) * DIMC + ci] = (_Float16)v;
    }
    __syncthreads();

    const int wave = t >> 5, lane = t & 31;
    const int coutBase = wave * 16;
    const int ln = lane & 15;
    const int hi = lane >> 4;                  // 0: lanes0-15, 1: lanes16-31

    v8f acc = {};
    for (int tap = 0; tap < 9; ++tap) {
        const int dy = tap / 3, dx = tap % 3;
        const _Float16* wt = wh + ((size_t)tap * DIMC + coutBase + ln) * DIMC;
        const _Float16* bp = &s_in[(dy * 18 + ln + dx) * DIMC];
        #pragma unroll
        for (int kc = 0; kc < 4; ++kc) {
            v16h a, bf;
            const int kbA = kc * 32 + hi * 8;          // A 16x32 f16 layout
            #pragma unroll
            for (int j = 0; j < 8; ++j) { a[j] = wt[kbA + j]; a[j + 8] = wt[kbA + 16 + j]; }
            const int kbB = kc * 32 + hi * 8;          // B 32x16 f16 layout
            #pragma unroll
            for (int j = 0; j < 8; ++j) { bf[j] = bp[kbB + j]; bf[j + 8] = bp[kbB + 16 + j]; }
            acc = __builtin_amdgcn_wmma_f32_16x16x32_f16(false, a, false, bf,
                                                         (short)0, acc, false, false);
        }
    }

    // C layout: VGPR r -> row (cout) = r + 8*hi ; col (pixel) = ln
    const int coBase = coutBase + hi * 8;
    __align__(16) _Float16 hv[8];
    #pragma unroll
    for (int r = 0; r < 8; ++r) hv[r] = (_Float16)(acc[r] + bias[coBase + r]);
    _Float16* op = out + (((size_t)b * HH + h) * WW + (w0 + ln)) * DIMC + coBase;
    *reinterpret_cast<uint4*>(op) = *reinterpret_cast<const uint4*>(hv);
}

// ---------------------------------------------------------------------------
__device__ __forceinline__ int shift_lab(int h, int s) {
    return s ? ((h < 256 - 2 * s) ? 0 : ((h < 256 - s) ? 1 : 2)) : 0;
}

// one WG (128 thr = 4 waves) per (window, part, estimate):
// async global->LDS window fill, corr GEMM + scale + rel-pos bias + shift mask
// -> LDS, then fused flow_bsd (center 4x4) and flow_mid (9x9 + quad softmax).
__global__ __launch_bounds__(128) void corrflow_kernel(const _Float16* __restrict__ q0h,
                                                       const _Float16* __restrict__ k0h,
                                                       const _Float16* __restrict__ q2h,
                                                       const _Float16* __restrict__ k2h,
                                                       const float* __restrict__ bias_table,
                                                       float* __restrict__ f1parts,
                                                       float* __restrict__ f0parts) {
    __shared__ __align__(16) _Float16 qs[64 * DIMC];
    __shared__ __align__(16) _Float16 ks[64 * DIMC];
    __shared__ float cr[64 * 64];
    __shared__ float midC[81], midFx[81], midFy[81];

    const int t    = threadIdx.x;
    const int part = blockIdx.y;                 // 0..3 shift combos
    const int est  = blockIdx.z;                 // 0:(q0,k2) 1:(q2,k0)
    const int b    = blockIdx.x >> 10;
    const int widx = blockIdx.x & 1023;
    const int s1i  = widx >> 5, s2i = widx & 31;
    const int sh = (part >> 1) ? (PW / 2) : 0;
    const int sw = (part & 1)  ? (PW / 2) : 0;

    const _Float16* qg = est ? q2h : q0h;
    const _Float16* kg = est ? k0h : k2h;

    // shifted 8x8 windows (NHWC f16, contiguous channels), copied straight
    // into LDS with CDNA5 async global->LDS B128 transfers (ASYNCcnt-tracked).
    {
        const int l  = t >> 1;
        const int c0 = (t & 1) * 64;             // 64 halves = 128 B per thread
        const int i = l >> 3, j = l & 7;
        const int hs = (s1i * PW + i + sh) & 255;
        const int wsp = (s2i * PW + j + sw) & 255;
        const size_t goff = (((size_t)b * HH + hs) * WW + wsp) * DIMC + c0;
        const _Float16* gq = qg + goff;
        const _Float16* gk = kg + goff;
        // low 32 bits of a generic shared-memory address == LDS byte offset
        unsigned lq = (unsigned)(size_t)(&qs[l * DIMC + c0]);
        unsigned lk = (unsigned)(size_t)(&ks[l * DIMC + c0]);
        #pragma unroll
        for (int m = 0; m < 8; ++m) {
            asm volatile("global_load_async_to_lds_b128 %0, %1, off"
                         :: "v"(lq + m * 16), "v"(gq + m * 8) : "memory");
            asm volatile("global_load_async_to_lds_b128 %0, %1, off"
                         :: "v"(lk + m * 16), "v"(gk + m * 8) : "memory");
        }
        asm volatile("s_wait_asynccnt 0" ::: "memory");
    }
    __syncthreads();

    // corr = qw . kw^T  (64x64, K=128) : 4 row-tiles x 4 col-tiles x 4 kchunks
    const int wave = t >> 5, lane = t & 31;
    const int ln = lane & 15, hi = lane >> 4;
    v8f acc[4] = {{}, {}, {}, {}};
    for (int kc = 0; kc < 4; ++kc) {
        v16h a;
        const _Float16* ap = &qs[(wave * 16 + ln) * DIMC + kc * 32 + hi * 8];
        #pragma unroll
        for (int j = 0; j < 8; ++j) { a[j] = ap[j]; a[j + 8] = ap[16 + j]; }
        #pragma unroll
        for (int u = 0; u < 4; ++u) {
            v16h bf;
            const _Float16* bp = &ks[(u * 16 + ln) * DIMC + kc * 32 + hi * 8];
            #pragma unroll
            for (int j = 0; j < 8; ++j) { bf[j] = bp[j]; bf[j + 8] = bp[16 + j]; }
            acc[u] = __builtin_amdgcn_wmma_f32_16x16x32_f16(false, a, false, bf,
                                                            (short)0, acc[u], false, false);
        }
    }
    // epilogue: scale + bias + shift mask -> LDS
    #pragma unroll
    for (int u = 0; u < 4; ++u) {
        #pragma unroll
        for (int r = 0; r < 8; ++r) {
            const int row = wave * 16 + hi * 8 + r;        // query l
            const int col = u * 16 + ln;                   // target t
            const int qi = row >> 3, qj = row & 7, ti = col >> 3, tj = col & 7;
            float v = acc[u][r] * SCALE_F +
                      bias_table[(qi - ti + 7) * 15 + (qj - tj + 7)];
            if (sh | sw) {
                const int lq = shift_lab(s1i * PW + qi, sh) * 3 + shift_lab(s2i * PW + qj, sw);
                const int lt = shift_lab(s1i * PW + ti, sh) * 3 + shift_lab(s2i * PW + tj, sw);
                if (lq != lt) v -= 10000.f;
            }
            cr[row * 64 + col] = v;
        }
    }
    __syncthreads();

    const size_t win = (size_t)b * (S1 * S2) + s1i * S2 + s2i;
    const size_t pslot = (size_t)est * 4 + part;

    if (t < 81) {
        // flow_mid stage 1: 9x9 transformed-grid soft-argmax
        const int hp = t / 9, wp = t % 9;
        float m = -1e30f;
        for (int k = 0; k < 64; ++k) {
            const int h2 = k >> 3, w2 = k & 7;
            const int h0 = 3 + hp - h2, w0 = 3 + wp - w2;
            const float v = (h0 >= 0 && h0 < 8 && w0 >= 0 && w0 < 8)
                                ? cr[(h0 * 8 + w0) * 64 + k] : 0.f;
            m = fmaxf(m, v);
        }
        float s = 0.f, cs = 0.f, fx = 0.f, fy = 0.f;
        const float bx = 0.5f * (float)(wp + 3), by = 0.5f * (float)(hp + 3);
        for (int k = 0; k < 64; ++k) {
            const int h2 = k >> 3, w2 = k & 7;
            const int h0 = 3 + hp - h2, w0 = 3 + wp - w2;
            const float v = (h0 >= 0 && h0 < 8 && w0 >= 0 && w0 < 8)
                                ? cr[(h0 * 8 + w0) * 64 + k] : 0.f;
            const float e = __expf(v - m);
            s += e; cs += v * e;
            fx += e * ((float)w2 - bx);
            fy += e * ((float)h2 - by);
        }
        const float inv = 1.f / s;
        midC[t] = cs * inv; midFx[t] = fx * inv; midFy[t] = fy * inv;
    } else if (t < 97) {
        // flow_bsd: 16 center queries, soft-argmax over 64 targets
        const int tb = t - 81;
        const int qi = 2 + (tb >> 2), qj = 2 + (tb & 3);
        const float* rowp = &cr[(qi * 8 + qj) * 64];
        float m = -1e30f;
        for (int k = 0; k < 64; ++k) m = fmaxf(m, rowp[k]);
        float s = 0.f, fx = 0.f, fy = 0.f;
        for (int k = 0; k < 64; ++k) {
            const float e = __expf(rowp[k] - m);
            s += e;
            fx += e * (float)((k & 7) - qj);
            fy += e * (float)((k >> 3) - qi);
        }
        const float inv = 1.f / s;
        float* o = f0parts + (pslot * NWIN + win) * 32;
        o[tb]      = fx * inv;
        o[16 + tb] = fy * inv;
    }
    __syncthreads();

    if (t < 64) {
        // flow_mid stage 2: 4-corner softmax blend, x2 scaling
        const int i = t >> 3, j = t & 7;
        const int n0 = i * 9 + j, n1 = n0 + 1, n2 = n0 + 9, n3 = n0 + 10;
        const float c0 = midC[n0], c1 = midC[n1], c2 = midC[n2], c3 = midC[n3];
        const float m = fmaxf(fmaxf(c0, c1), fmaxf(c2, c3));
        const float e0 = __expf(c0 - m), e1 = __expf(c1 - m);
        const float e2 = __expf(c2 - m), e3 = __expf(c3 - m);
        const float inv = 2.f / (e0 + e1 + e2 + e3);        // includes *2.0 factor
        float* o = f1parts + (pslot * NWIN + win) * 128;
        o[t]      = (midFx[n0] * e0 + midFx[n1] * e1 + midFx[n2] * e2 + midFx[n3] * e3) * inv;
        o[64 + t] = (midFy[n0] * e0 + midFy[n1] * e1 + midFy[n2] * e2 + midFy[n3] * e3) * inv;
    }
}

// ---------------------------------------------------------------------------
// splice 4 parts of flow_mid (8x8 each) into (2,2,512,512) with roll(+4,+4)
__global__ __launch_bounds__(256) void splice1_kernel(const float* __restrict__ f1parts,
                                                      float* __restrict__ out) {
    const int est = blockIdx.y;
    const int idx = blockIdx.x * 256 + threadIdx.x;   // < 2*2*512*512
    const int x = idx & 511, y = (idx >> 9) & 511, ch = idx >> 18;
    const int b = ch >> 1, d = ch & 1;
    const int yp = (y + 508) & 511, xp = (x + 508) & 511;   // roll by +4
    const int s1i = yp >> 4, s2i = xp >> 4, yy = yp & 15, xx = xp & 15;
    const int part = ((yy >> 3) << 1) | (xx >> 3);
    const size_t win = (size_t)b * 1024 + s1i * 32 + s2i;
    out[(size_t)est * 1048576 + idx] =
        f1parts[(((size_t)est * 4 + part) * NWIN + win) * 128 + d * 64 + (yy & 7) * 8 + (xx & 7)];
}

// splice 4 parts of flow_bsd (4x4 each) into (2,2,256,256) with roll(+2,+2)
__global__ __launch_bounds__(256) void splice0_kernel(const float* __restrict__ f0parts,
                                                      float* __restrict__ out) {
    const int est = blockIdx.y;
    const int idx = blockIdx.x * 256 + threadIdx.x;   // < 2*2*256*256
    const int x = idx & 255, y = (idx >> 8) & 255, ch = idx >> 16;
    const int b = ch >> 1, d = ch & 1;
    const int yp = (y + 254) & 255, xp = (x + 254) & 255;   // roll by +2
    const int s1i = yp >> 3, s2i = xp >> 3, yy = yp & 7, xx = xp & 7;
    const int part = ((yy >> 2) << 1) | (xx >> 2);
    const size_t win = (size_t)b * 1024 + s1i * 32 + s2i;
    out[(size_t)est * 262144 + idx] =
        f0parts[(((size_t)est * 4 + part) * NWIN + win) * 32 + d * 16 + (yy & 3) * 4 + (xx & 3)];
}

// ---------------------------------------------------------------------------
// bilinear resize (half-pixel centers) (2,2,sh,sw) -> (2,2,1024,1024), *mul
__global__ __launch_bounds__(256) void resize_kernel(const float* __restrict__ src,
                                                     float* __restrict__ dst,
                                                     int sh, int sw, float f, float mul) {
    const int idx = blockIdx.x * 256 + threadIdx.x;   // < 2*2*1024*1024
    const int x = idx & 1023, y = (idx >> 10) & 1023, ch = idx >> 20;
    float sy = (y + 0.5f) / f - 0.5f;
    float sx = (x + 0.5f) / f - 0.5f;
    sy = fminf(fmaxf(sy, 0.f), (float)(sh - 1));
    sx = fminf(fmaxf(sx, 0.f), (float)(sw - 1));
    const int y0 = (int)sy, x0 = (int)sx;
    const int y1 = min(y0 + 1, sh - 1), x1 = min(x0 + 1, sw - 1);
    const float wy = sy - (float)y0, wx = sx - (float)x0;
    const float* p = src + (size_t)ch * sh * sw;
    const float v = (1.f - wy) * ((1.f - wx) * p[y0 * sw + x0] + wx * p[y0 * sw + x1]) +
                    wy * ((1.f - wx) * p[y1 * sw + x0] + wx * p[y1 * sw + x1]);
    dst[idx] = v * mul;
}

// ---------------------------------------------------------------------------
extern "C" void kernel_launch(void* const* d_in, const int* in_sizes, int n_in,
                              void* d_out, int out_size, void* d_ws, size_t ws_size,
                              hipStream_t stream) {
    const float* feat0      = (const float*)d_in[0];
    const float* feat2      = (const float*)d_in[1];
    const float* wq         = (const float*)d_in[2];
    const float* bq         = (const float*)d_in[3];
    const float* wk         = (const float*)d_in[4];
    const float* bk         = (const float*)d_in[5];
    const float* bias_table = (const float*)d_in[6];
    float* out = (float*)d_out;

    // workspace carve-up (~156 MB)
    _Float16* q0h = (_Float16*)d_ws;                 // 2*128*256*256 halves each
    _Float16* k0h = q0h + 16777216;
    _Float16* q2h = k0h + 16777216;
    _Float16* k2h = q2h + 16777216;
    _Float16* wqh = k2h + 16777216;                  // 147456 halves
    _Float16* wkh = wqh + 147456;
    float* f1parts = (float*)(wkh + 147456);         // 2*4*2048*128 f32
    float* f0parts = f1parts + 2097152;              // 2*4*2048*32  f32
    float* spl1    = f0parts + 524288;               // 2 * (2,2,512,512)
    float* spl0    = spl1 + 2097152;                 // 2 * (2,2,256,256)

    pack_w_kernel<<<576, 256, 0, stream>>>(wq, wqh);
    pack_w_kernel<<<576, 256, 0, stream>>>(wk, wkh);

    const dim3 cg(16, 256, 2);
    conv3x3_kernel<<<cg, 256, 0, stream>>>(feat0, wqh, bq, q0h);
    conv3x3_kernel<<<cg, 256, 0, stream>>>(feat0, wkh, bk, k0h);
    conv3x3_kernel<<<cg, 256, 0, stream>>>(feat2, wqh, bq, q2h);
    conv3x3_kernel<<<cg, 256, 0, stream>>>(feat2, wkh, bk, k2h);

    corrflow_kernel<<<dim3(2048, 4, 2), 128, 0, stream>>>(q0h, k0h, q2h, k2h,
                                                          bias_table, f1parts, f0parts);

    splice1_kernel<<<dim3(4096, 2), 256, 0, stream>>>(f1parts, spl1);
    splice0_kernel<<<dim3(1024, 2), 256, 0, stream>>>(f0parts, spl0);

    // output order: flow10, flow12, flow02, flow20 — each (2,2,1024,1024)
    resize_kernel<<<16384, 256, 0, stream>>>(spl1 + 1048576, out + 0,        512, 512, 2.f, 2.f); // flow10 (est1 mid)
    resize_kernel<<<16384, 256, 0, stream>>>(spl1,           out + 4194304,  512, 512, 2.f, 2.f); // flow12 (est0 mid)
    resize_kernel<<<16384, 256, 0, stream>>>(spl0,           out + 8388608,  256, 256, 4.f, 4.f); // flow02 (est0 bsd)
    resize_kernel<<<16384, 256, 0, stream>>>(spl0 + 262144,  out + 12582912, 256, 256, 4.f, 4.f); // flow20 (est1 bsd)
}
